// GNNMinCut_807453851818
// MI455X (gfx1250) — matrix operand order repaired and checked by reading, not compiled, not run on hardware
//
#include <hip/hip_runtime.h>

// ---------------------------------------------------------------------------
// GNN MinCut on MI455X (gfx1250).
// Math identity: softmax assignment rows sum to 1 => mincut pool + cluster
// mean collapses to (1/K) * per-graph mean of h2; Wa/ba/S are dead.
// Work: 2x GCN(128->128) [WMMA f16->f32 GEMM + dst-CSR aggregation] +
// block mean + [50x128]@[128x10].
// Round-4 change: sched_barrier pins the 4 preloaded A fragments so the
// backend cannot sink the A loads back between WMMAs (removes mid-chain
// s_wait_loadcnt 0x0 serialization and WAR v_nops).
// ---------------------------------------------------------------------------

typedef __attribute__((ext_vector_type(16))) _Float16 v16h;
typedef __attribute__((ext_vector_type(8)))  float    v8f;

#if defined(__has_builtin)
#if __has_builtin(__builtin_amdgcn_sched_barrier)
#define SCHED_FENCE() __builtin_amdgcn_sched_barrier(0)
#endif
#endif
#ifndef SCHED_FENCE
#define SCHED_FENCE() ((void)0)
#endif

#define GN_N 100000
#define GN_B 50
#define GN_P 2000
#define GN_E 1600000
#define GN_F 128
#define GN_K 30
#define GN_C 10
#define SCAN_BLOCKS ((GN_N + 255) / 256)

// ---------------- small utility kernels ----------------

__global__ __launch_bounds__(256) void zero_int_kernel(int* p, int n) {
    int i = blockIdx.x * 256 + threadIdx.x;
    if (i < n) p[i] = 0;
}

__global__ __launch_bounds__(256) void deg_count_kernel(const int* __restrict__ dst,
                                                        int* __restrict__ deg, int e) {
    int i = blockIdx.x * 256 + threadIdx.x;
    if (i < e) atomicAdd(&deg[dst[i]], 1);
}

__global__ __launch_bounds__(256) void invsqrt_kernel(const int* __restrict__ deg,
                                                      float* __restrict__ isq, int n) {
    int i = blockIdx.x * 256 + threadIdx.x;
    if (i < n) isq[i] = rsqrtf((float)deg[i] + 1.0f);   // +1 self loop
}

// ---------------- exclusive scan of degrees (3 kernels) ----------------

__global__ __launch_bounds__(256) void block_sum_kernel(const int* __restrict__ deg,
                                                        int* __restrict__ bsum, int n) {
    __shared__ int s[256];
    int i = blockIdx.x * 256 + threadIdx.x;
    s[threadIdx.x] = (i < n) ? deg[i] : 0;
    __syncthreads();
    for (int off = 128; off > 0; off >>= 1) {
        if (threadIdx.x < off) s[threadIdx.x] += s[threadIdx.x + off];
        __syncthreads();
    }
    if (threadIdx.x == 0) bsum[blockIdx.x] = s[0];
}

__global__ void scan_sums_kernel(int* bsum, int nb) {
    if (blockIdx.x == 0 && threadIdx.x == 0) {
        int run = 0;
        for (int i = 0; i < nb; ++i) { int t = bsum[i]; bsum[i] = run; run += t; }
    }
}

__global__ __launch_bounds__(256) void scan_write_kernel(const int* __restrict__ deg,
                                                         const int* __restrict__ boff,
                                                         int* __restrict__ row_start,
                                                         int* __restrict__ cursor,
                                                         int n, int total) {
    __shared__ int s[256];
    int i = blockIdx.x * 256 + threadIdx.x;
    int v = (i < n) ? deg[i] : 0;
    s[threadIdx.x] = v;
    __syncthreads();
    for (int off = 1; off < 256; off <<= 1) {          // Hillis-Steele inclusive
        int tmp = (threadIdx.x >= off) ? s[threadIdx.x - off] : 0;
        __syncthreads();
        s[threadIdx.x] += tmp;
        __syncthreads();
    }
    int excl = s[threadIdx.x] - v;
    if (i < n) {
        int rs = excl + boff[blockIdx.x];
        row_start[i] = rs;
        cursor[i]    = rs;
    }
    if (i == 0) row_start[n] = total;
}

__global__ __launch_bounds__(256) void bucket_kernel(const int* __restrict__ src,
                                                     const int* __restrict__ dst,
                                                     const float* __restrict__ isq,
                                                     int* __restrict__ cursor,
                                                     int* __restrict__ col,
                                                     float* __restrict__ coefv, int e) {
    int i = blockIdx.x * 256 + threadIdx.x;
    if (i >= e) return;
    int d = dst[i], s = src[i];
    int pos = atomicAdd(&cursor[d], 1);
    col[pos]   = s;
    coefv[pos] = isq[s] * isq[d];
}

// ---------------- pre-swizzle W into WMMA B-fragment layout ----------------
// B frag (16x16x32 f16, wave32): lane n(0-15) holds column n, K=0..7,16..23;
// lanes 16-31 hold K=8..15,24..31 (matches A fragment k order).
// Stored: frag index (kt*8+nt), then lane, then 16 halfs contiguous.

__global__ __launch_bounds__(256) void wswz_kernel(const float* __restrict__ W,
                                                   _Float16* __restrict__ dstp) {
    int t = blockIdx.x * 256 + threadIdx.x;            // [0,1024)
    if (t >= 4 * 8 * 32) return;
    int lane = t & 31;
    int nt   = (t >> 5) & 7;
    int kt   = t >> 8;
    int n  = nt * 16 + (lane & 15);
    int k0 = kt * 32 + (lane >> 4) * 8;
    _Float16* d = dstp + (size_t)t * 16;
    #pragma unroll
    for (int j = 0; j < 16; ++j) d[j] = (_Float16)W[(size_t)(k0 + (j < 8 ? j : j + 8)) * GN_F + n];
}

// ---------------- WMMA GEMM: Y[N,128] = X[N,128] @ W[128,128] ----------------
// B fragments (32 KB) staged once per block in LDS. 1 wave -> 16x128 strip.
// All 4 A fragments preloaded (one load clause + converts), pinned by a
// sched_barrier, then a clean 32x v_wmma_f32_16x16x32_f16 chain with B
// streaming from LDS.

__global__ __launch_bounds__(256) void gemm_wmma_kernel(const float* __restrict__ X,
                                                        const v16h* __restrict__ Wswz,
                                                        float* __restrict__ Y,
                                                        int nrows) {
    __shared__ v16h ldsW[32 * 32];                     // 32 frags x 32 lanes = 32 KB

    // cooperative stage: 32 KB = 2048 float4, 8 per thread
    {
        const float4* wsrc = (const float4*)Wswz;
        float4* wdst = (float4*)ldsW;
        #pragma unroll
        for (int i = 0; i < 8; ++i)
            wdst[threadIdx.x + 256 * i] = wsrc[threadIdx.x + 256 * i];
    }
    __syncthreads();                                   // all waves hit the barrier

    int lane = threadIdx.x & 31;
    int wave = threadIdx.x >> 5;
    int r0 = (blockIdx.x * 8 + wave) * 16;
    if (r0 >= nrows) return;                           // wave-uniform, after barrier
    int lm = lane & 15;
    int hs = lane >> 4;                                // 0 or 1
    int m  = r0 + lm;
    int mc = (m < nrows) ? m : (nrows - 1);
    const float* xrow = X + (size_t)mc * GN_F;

    // preload + convert all 4 A fragments (ISA 7.12.2 16-bit A 16x32 layout)
    v16h a[4];
    #pragma unroll
    for (int kt = 0; kt < 4; ++kt) {
        int kb = kt * 32 + hs * 8;
        float4 f0 = *(const float4*)(xrow + kb);
        float4 f1 = *(const float4*)(xrow + kb + 4);
        float4 f2 = *(const float4*)(xrow + kb + 16);
        float4 f3 = *(const float4*)(xrow + kb + 20);
        a[kt][0]  = (_Float16)f0.x; a[kt][1]  = (_Float16)f0.y;
        a[kt][2]  = (_Float16)f0.z; a[kt][3]  = (_Float16)f0.w;
        a[kt][4]  = (_Float16)f1.x; a[kt][5]  = (_Float16)f1.y;
        a[kt][6]  = (_Float16)f1.z; a[kt][7]  = (_Float16)f1.w;
        a[kt][8]  = (_Float16)f2.x; a[kt][9]  = (_Float16)f2.y;
        a[kt][10] = (_Float16)f2.z; a[kt][11] = (_Float16)f2.w;
        a[kt][12] = (_Float16)f3.x; a[kt][13] = (_Float16)f3.y;
        a[kt][14] = (_Float16)f3.z; a[kt][15] = (_Float16)f3.w;
    }

    // keep the A fragments resident: forbid sinking the loads into the chain
    SCHED_FENCE();

    v8f acc[8];
    #pragma unroll
    for (int i = 0; i < 8; ++i) {
        v8f z = {0.f, 0.f, 0.f, 0.f, 0.f, 0.f, 0.f, 0.f};
        acc[i] = z;
    }

    // 32 WMMAs; consecutive ops hit different accumulators -> independent
    #pragma unroll
    for (int kt = 0; kt < 4; ++kt) {
        #pragma unroll
        for (int nt = 0; nt < 8; ++nt) {
            v16h b = ldsW[(kt * 8 + nt) * 32 + lane];  // ds_load, short latency
            acc[nt] = __builtin_amdgcn_wmma_f32_16x16x32_f16(
                false, a[kt], false, b, (short)0, acc[nt], false, false);
        }
    }

    // C/D layout: VGPR j -> row r0 + hs*8 + j, col nt*16 + lm
    int mbase = r0 + hs * 8;
    #pragma unroll
    for (int nt = 0; nt < 8; ++nt) {
        #pragma unroll
        for (int j = 0; j < 8; ++j) {
            int mm = mbase + j;
            if (mm < nrows) Y[(size_t)mm * GN_F + nt * 16 + lm] = acc[nt][j];
        }
    }
}

// ---------------- CSR aggregate + self-loop + bias (+ReLU) ----------------
// 1 wave per node row; lane owns 4 features (float4). Edge metadata is loaded
// coalesced in chunks of 32 and broadcast across the wave with __shfl, so the
// only per-edge VMEM op is the 512B feature-row gather (L2-resident).

__global__ __launch_bounds__(256) void aggregate_kernel(const float* __restrict__ Y,
                                                        const int* __restrict__ row_start,
                                                        const int* __restrict__ col,
                                                        const float* __restrict__ coefv,
                                                        const float* __restrict__ isq,
                                                        const float* __restrict__ bias,
                                                        float* __restrict__ H,
                                                        int n, int do_relu) {
    int lane = threadIdx.x & 31;
    int row  = blockIdx.x * 8 + (threadIdx.x >> 5);
    if (row >= n) return;
    const float4* yrow = (const float4*)(Y + (size_t)row * GN_F);
    float4 self = yrow[lane];
    float4 acc = make_float4(0.f, 0.f, 0.f, 0.f);
    int e0 = row_start[row], e1 = row_start[row + 1];
    for (int base = e0; base < e1; base += 32) {
        int idx = base + lane;
        int   cl = (idx < e1) ? col[idx]   : 0;        // coalesced 128B
        float wl = (idx < e1) ? coefv[idx] : 0.f;      // coalesced 128B
        int cnt = e1 - base; if (cnt > 32) cnt = 32;
        for (int j = 0; j < cnt; ++j) {
            int   c = __shfl(cl, j, 32);
            float w = __shfl(wl, j, 32);
            float4 v = ((const float4*)(Y + (size_t)c * GN_F))[lane];
            acc.x += w * v.x; acc.y += w * v.y; acc.z += w * v.z; acc.w += w * v.w;
        }
    }
    float iv = isq[row];
    float i2 = iv * iv;
    float4 bv = ((const float4*)bias)[lane];
    float4 o;
    o.x = acc.x + self.x * i2 + bv.x;
    o.y = acc.y + self.y * i2 + bv.y;
    o.z = acc.z + self.z * i2 + bv.z;
    o.w = acc.w + self.w * i2 + bv.w;
    if (do_relu) {
        o.x = fmaxf(o.x, 0.f); o.y = fmaxf(o.y, 0.f);
        o.z = fmaxf(o.z, 0.f); o.w = fmaxf(o.w, 0.f);
    }
    ((float4*)(H + (size_t)row * GN_F))[lane] = o;
}

// ---------------- per-graph mean (/K) + logits ----------------

__global__ __launch_bounds__(256) void pool_logits_kernel(const float* __restrict__ H,
                                                          const float* __restrict__ Wl,
                                                          const float* __restrict__ bl,
                                                          float* __restrict__ out) {
    __shared__ float s[256];
    __shared__ float pool[GN_F];
    int b = blockIdx.x;
    int t = threadIdx.x;
    int f = t & 127;
    int half = t >> 7;
    const float* base = H + (size_t)b * GN_P * GN_F;
    float sum = 0.f;
    for (int nrow = half; nrow < GN_P; nrow += 2) sum += base[(size_t)nrow * GN_F + f];
    s[t] = sum;
    __syncthreads();
    if (t < GN_F) pool[t] = (s[t] + s[t + 128]) * (1.0f / (float)GN_K);
    __syncthreads();
    if (t < GN_C) {
        float acc = bl[t];
        for (int d = 0; d < GN_F; ++d) acc += pool[d] * Wl[d * GN_C + t];
        out[b * GN_C + t] = acc;
    }
}

// ---------------- launcher ----------------

extern "C" void kernel_launch(void* const* d_in, const int* in_sizes, int n_in,
                              void* d_out, int out_size, void* d_ws, size_t ws_size,
                              hipStream_t stream) {
    (void)in_sizes; (void)n_in; (void)out_size; (void)ws_size;
    const float* x  = (const float*)d_in[0];
    const int*   ei = (const int*)d_in[1];
    const int*   src = ei;
    const int*   dst = ei + GN_E;
    const float* W1 = (const float*)d_in[3];
    const float* b1 = (const float*)d_in[4];
    const float* W2 = (const float*)d_in[5];
    const float* b2 = (const float*)d_in[6];
    // d_in[7]=Wa, d_in[8]=ba: provably dead (softmax rows sum to 1)
    const float* Wl = (const float*)d_in[9];
    const float* bl = (const float*)d_in[10];
    float* out = (float*)d_out;

    // workspace carve (~118 MB)
    char* p = (char*)d_ws;
    auto carve = [&](size_t bytes) -> void* {
        void* r = (void*)p;
        p += (bytes + 255) & ~(size_t)255;
        return r;
    };
    int*      deg       = (int*)carve((size_t)GN_N * 4);
    float*    isq       = (float*)carve((size_t)GN_N * 4);
    int*      row_start = (int*)carve((size_t)(GN_N + 1) * 4);
    int*      cursor    = (int*)carve((size_t)GN_N * 4);
    int*      bsum      = (int*)carve((size_t)SCAN_BLOCKS * 4);
    int*      col       = (int*)carve((size_t)GN_E * 4);
    float*    coefv     = (float*)carve((size_t)GN_E * 4);
    float*    Ybuf      = (float*)carve((size_t)GN_N * GN_F * 4);
    float*    Hbuf      = (float*)carve((size_t)GN_N * GN_F * 4);
    _Float16* Wswz1     = (_Float16*)carve((size_t)4 * 8 * 32 * 16 * 2);
    _Float16* Wswz2     = (_Float16*)carve((size_t)4 * 8 * 32 * 16 * 2);

    const int gN = (GN_N + 255) / 256;
    const int gE = (GN_E + 255) / 256;

    // CSR build
    zero_int_kernel<<<gN, 256, 0, stream>>>(deg, GN_N);
    deg_count_kernel<<<gE, 256, 0, stream>>>(dst, deg, GN_E);
    invsqrt_kernel<<<gN, 256, 0, stream>>>(deg, isq, GN_N);
    block_sum_kernel<<<SCAN_BLOCKS, 256, 0, stream>>>(deg, bsum, GN_N);
    scan_sums_kernel<<<1, 64, 0, stream>>>(bsum, SCAN_BLOCKS);
    scan_write_kernel<<<SCAN_BLOCKS, 256, 0, stream>>>(deg, bsum, row_start, cursor, GN_N, GN_E);
    bucket_kernel<<<gE, 256, 0, stream>>>(src, dst, isq, cursor, col, coefv, GN_E);

    // weight pre-swizzle into WMMA B-fragment layout
    wswz_kernel<<<4, 256, 0, stream>>>(W1, Wswz1);
    wswz_kernel<<<4, 256, 0, stream>>>(W2, Wswz2);

    const int gGemm = (GN_N + 127) / 128;   // 8 waves x 16 rows per block
    const int gAgg  = (GN_N + 7) / 8;       // 8 waves (rows) per block

    // layer 1
    gemm_wmma_kernel<<<gGemm, 256, 0, stream>>>(x, (const v16h*)Wswz1, Ybuf, GN_N);
    aggregate_kernel<<<gAgg, 256, 0, stream>>>(Ybuf, row_start, col, coefv, isq, b1, Hbuf, GN_N, 1);
    // layer 2 (ping-pong: Y reused, H overwritten after its last read)
    gemm_wmma_kernel<<<gGemm, 256, 0, stream>>>(Hbuf, (const v16h*)Wswz2, Ybuf, GN_N);
    aggregate_kernel<<<gAgg, 256, 0, stream>>>(Ybuf, row_start, col, coefv, isq, b2, Hbuf, GN_N, 1);

    // pooled mean (/K) + classifier
    pool_logits_kernel<<<GN_B, 256, 0, stream>>>(Hbuf, Wl, bl, out);
}
